// MultiHeadAttention_22419729285517
// MI455X (gfx1250) — matrix operand order repaired and checked by reading, not compiled, not run on hardware
//
#include <hip/hip_runtime.h>

// ---------------------------------------------------------------------------
// Problem constants (B=2, N=4096, C=768, H=12, D=64, 3C=2304)
// ---------------------------------------------------------------------------
#define BB   2
#define NN   4096
#define CC   768
#define HH   12
#define DD   64
#define TC   2304
#define ROWS 8192          // B*N
#define LDT  40            // padded LDS row length (bf16 elems) for 32-wide tiles
#define TBUF (128 * LDT)   // elements per LDS tile buffer

typedef __bf16        v16bf __attribute__((ext_vector_type(16)));
typedef float         v8f   __attribute__((ext_vector_type(8)));
typedef unsigned int  v4u   __attribute__((ext_vector_type(4)));
typedef unsigned int  u32x4 __attribute__((ext_vector_type(4)));
typedef int           i32x4 __attribute__((ext_vector_type(4)));
typedef int           i32x8 __attribute__((ext_vector_type(8)));

union ABFrag { v16bf v; v4u q[2]; };

#if defined(__HIP_DEVICE_COMPILE__) && defined(__gfx1250__)
#define ON_GFX1250 1
#else
#define ON_GFX1250 0
#endif

#if ON_GFX1250 && __has_builtin(__builtin_amdgcn_tensor_load_to_lds)
#define HAVE_TDM 1
#else
#define HAVE_TDM 0
#endif

__device__ __forceinline__ unsigned short f2bf(float f) {
  unsigned int u = __builtin_bit_cast(unsigned int, f);
  u += 0x7fffu + ((u >> 16) & 1u);            // round-to-nearest-even
  return (unsigned short)(u >> 16);
}

__device__ __forceinline__ v8f wmma_bf16(const ABFrag& a, const ABFrag& b, v8f c) {
  return __builtin_amdgcn_wmma_f32_16x16x32_bf16(false, a.v, false, b.v,
                                                 (short)0, c, false, false);
}

// --- CDNA5 async global->LDS copy (ASYNCcnt) --------------------------------
__device__ __forceinline__ void async_b128(const void* g, void* l) {
#if ON_GFX1250
  unsigned lds = (unsigned)(size_t)l;                 // low 32 bits = LDS offset
  unsigned long long ga = (unsigned long long)(size_t)g;
  asm volatile("global_load_async_to_lds_b128 %0, %1, off"
               :: "v"(lds), "v"(ga) : "memory");
#else
  *(v4u*)l = *(const v4u*)g;
#endif
}

__device__ __forceinline__ void wait_async0() {
#if ON_GFX1250
  asm volatile("s_wait_asynccnt 0x0" ::: "memory");
#endif
}

// --- CDNA5 Tensor Data Mover: 2-D tile load with LDS padding (TENSORcnt) ----
#if HAVE_TDM
// pad: insert 4 DWORDs (code 3) after every 32 DWORDs (code 4) -> 128B rows
// land at a 144B LDS stride (72 bf16), matching the compute-side layout.
__device__ __forceinline__ void tdm_load_2d_pad(
    unsigned lds_addr, const void* gaddr,
    unsigned dim0, unsigned dim1,      // tensor dims (elements), dim0 = row len
    unsigned tile0, unsigned tile1,    // tile dims (elements)
    unsigned stride0)                  // row stride (elements)
{
  unsigned long long ga = (unsigned long long)(size_t)gaddr;
  u32x4 g0;
  g0[0] = 1u;                                        // count=1, user descriptor
  g0[1] = lds_addr;                                  // lds_addr [63:32]
  g0[2] = (unsigned)(ga & 0xffffffffu);              // global_addr lo
  g0[3] = (unsigned)((ga >> 32) & 0x01ffffffu) | 0x80000000u; // addr hi | type=2
  i32x8 g1;
  g1[0] = (int)((1u << 16)        // data_size = 1 (2 bytes)
              | (1u << 20)        // pad_enable
              | (4u << 22)        // pad_interval: 32 DWORDs
              | (3u << 25));      // pad_amount: 4 DWORDs
  g1[1] = (int)((dim0 & 0xffffu) << 16);             // tensor_dim0 lo (bar addr 0)
  g1[2] = (int)((dim0 >> 16) | ((dim1 & 0xffffu) << 16));
  g1[3] = (int)((dim1 >> 16) | (tile0 << 16));       // tile_dim0
  g1[4] = (int)(tile1 & 0xffffu);                    // tile_dim1 (tile_dim2 = 0)
  g1[5] = (int)stride0;                              // tensor_dim0_stride lo
  g1[6] = 0;                                         // stride hi / dim1_stride
  g1[7] = 0;
  i32x4 z4 = {0, 0, 0, 0};
#if __clang_major__ >= 23
  i32x8 z8 = {0, 0, 0, 0, 0, 0, 0, 0};
  __builtin_amdgcn_tensor_load_to_lds(g0, g1, z4, z4, z8, 0);
#else
  __builtin_amdgcn_tensor_load_to_lds(g0, g1, z4, z4, 0);
#endif
}
#endif

// ---------------------------------------------------------------------------
// Shared 128x128 GEMM main loop (bf16 A [M x K] row-major, bf16 B^T [N x K]
// row-major, f32 accum). 8 waves; wave tile 64x32. Double-buffered LDS fed by
// async global->LDS copies: prefetch k+1 while WMMAs run on k.
// ---------------------------------------------------------------------------
__device__ __forceinline__ void gemm128x128_bf16(
    const unsigned short* __restrict__ A, int lda,
    const unsigned short* __restrict__ BT, int ldb,
    int m0, int n0, int Kdim,
    unsigned short* ldsA, unsigned short* ldsB,   // each 2 * TBUF elements
    v8f (&acc)[4][2])
{
  const int t    = threadIdx.x;
  const int lane = t & 31, w = t >> 5;
  const int wm   = w & 1,  wn = w >> 1;
  const int h    = lane >> 4, r = lane & 15;
  const int row  = t >> 1, part = t & 1;

  const unsigned short* gA = A  + (size_t)(m0 + row) * lda + part * 16;
  const unsigned short* gB = BT + (size_t)(n0 + row) * ldb + part * 16;
  unsigned short* lA = ldsA + row * LDT + part * 16;
  unsigned short* lB = ldsB + row * LDT + part * 16;

  // prologue: tile k0=0 into buffer 0
  async_b128(gA,     lA);      async_b128(gA + 8, lA + 8);
  async_b128(gB,     lB);      async_b128(gB + 8, lB + 8);
  wait_async0();
  __syncthreads();

  int cur = 0;
  for (int k0 = 0; k0 < Kdim; k0 += 32) {
    const int nxt = cur ^ 1;
    if (k0 + 32 < Kdim) {       // prefetch next k-tile into the other buffer
      async_b128(gA + k0 + 32,     lA + nxt * TBUF);
      async_b128(gA + k0 + 32 + 8, lA + nxt * TBUF + 8);
      async_b128(gB + k0 + 32,     lB + nxt * TBUF);
      async_b128(gB + k0 + 32 + 8, lB + nxt * TBUF + 8);
    }
    const unsigned short* bA = ldsA + cur * TBUF;
    const unsigned short* bB = ldsB + cur * TBUF;

    ABFrag bfr[2];
#pragma unroll
    for (int nt = 0; nt < 2; ++nt) {
      const unsigned short* p = bB + (wn * 32 + nt * 16 + r) * LDT + 16 * h;
      bfr[nt].q[0] = *(const v4u*)p;
      bfr[nt].q[1] = *(const v4u*)(p + 8);
    }
#pragma unroll
    for (int mt = 0; mt < 4; ++mt) {
      ABFrag af;
      const unsigned short* p = bA + (wm * 64 + mt * 16 + r) * LDT + 16 * h;
      af.q[0] = *(const v4u*)p;
      af.q[1] = *(const v4u*)(p + 8);
#pragma unroll
      for (int nt = 0; nt < 2; ++nt)
        acc[mt][nt] = wmma_bf16(af, bfr[nt], acc[mt][nt]);
    }
    wait_async0();
    __syncthreads();
    cur = nxt;
  }
}

// ---------------------------------------------------------------------------
// Conversion kernels (run once per call; tiny)
// ---------------------------------------------------------------------------
__global__ void cvt_f32_bf16(const float* __restrict__ in,
                             unsigned short* __restrict__ out, int n) {
  int i = blockIdx.x * blockDim.x + threadIdx.x;
  if (i < n) out[i] = f2bf(in[i]);
}

__global__ void cvt_transpose_bf16(const float* __restrict__ in,
                                   unsigned short* __restrict__ out,
                                   int R, int Ccols) {
  int i = blockIdx.x * blockDim.x + threadIdx.x;
  if (i < R * Ccols) {
    int c = i / R, r = i % R;
    out[i] = f2bf(in[(size_t)r * Ccols + c]);
  }
}

// ---------------------------------------------------------------------------
// K1: QKV GEMM  -> Q [B,H,N,D], Kt/Vt [B,H,D,N]
// ---------------------------------------------------------------------------
__global__ __launch_bounds__(256) void qkv_gemm(
    const unsigned short* __restrict__ Xb,      // [8192][768] bf16
    const unsigned short* __restrict__ WqkvT,   // [2304][768] bf16
    const float* __restrict__ b_qkv,            // [2304]
    unsigned short* __restrict__ Qb,            // [B,H,N,64]
    unsigned short* __restrict__ Kt,            // [B,H,64,N]
    unsigned short* __restrict__ Vt)            // [B,H,64,N]
{
  __shared__ unsigned short ldsA[2 * TBUF];
  __shared__ unsigned short ldsB[2 * TBUF];
  const int m0 = blockIdx.x * 128, n0 = blockIdx.y * 128;
  v8f acc[4][2] = {};
  gemm128x128_bf16(Xb, CC, WqkvT, CC, m0, n0, CC, ldsA, ldsB, acc);

  const int lane = threadIdx.x & 31, w = threadIdx.x >> 5;
  const int wm = w & 1, wn = w >> 1;
  const int h = lane >> 4, r = lane & 15;
#pragma unroll
  for (int mt = 0; mt < 4; ++mt) {
#pragma unroll
    for (int nt = 0; nt < 2; ++nt) {
      const int col   = n0 + wn * 32 + nt * 16 + r;
      const float bias = b_qkv[col];
      const int which = col / CC, cc = col % CC;
      const int head = cc >> 6, d = cc & 63;
#pragma unroll
      for (int v = 0; v < 8; ++v) {
        const int row = m0 + wm * 64 + mt * 16 + v + 8 * h;
        const int b = row >> 12, n = row & 4095;
        const unsigned short bf = f2bf(acc[mt][nt][v] + bias);
        const size_t bh = (size_t)b * HH + head;
        if (which == 0)      Qb[(bh * NN + n) * DD + d] = bf;
        else if (which == 1) Kt[(bh * DD + d) * NN + n] = bf;
        else                 Vt[(bh * DD + d) * NN + n] = bf;
      }
    }
  }
}

// ---------------------------------------------------------------------------
// K2: M[b,h] = (k^T v) / 8  -> stored transposed MT [B,H, d, e] bf16
// ---------------------------------------------------------------------------
__global__ __launch_bounds__(256) void kv_outer(
    const unsigned short* __restrict__ Kt,   // [B,H,64,N]
    const unsigned short* __restrict__ Vt,   // [B,H,64,N]
    unsigned short* __restrict__ MTb)        // [B,H,64(d),64(e)]
{
  __shared__ float sred[4 * 4096];           // 64 KB
  const int bh = blockIdx.x;
  const unsigned short* Kp = Kt + (size_t)bh * DD * NN;
  const unsigned short* Vp = Vt + (size_t)bh * DD * NN;
  const int t = threadIdx.x, lane = t & 31, w = t >> 5;
  const int h = lane >> 4, r = lane & 15;

  v8f acc[4][4] = {};
  for (int it = 0; it < 16; ++it) {
    const int n0 = w * 512 + it * 32;
    ABFrag bf[4];
#pragma unroll
    for (int dt = 0; dt < 4; ++dt) {
      const unsigned short* p = Vp + (size_t)(dt * 16 + r) * NN + n0 + 16 * h;
      bf[dt].q[0] = *(const v4u*)p;
      bf[dt].q[1] = *(const v4u*)(p + 8);
    }
#pragma unroll
    for (int et = 0; et < 4; ++et) {
      ABFrag af;
      const unsigned short* p = Kp + (size_t)(et * 16 + r) * NN + n0 + 16 * h;
      af.q[0] = *(const v4u*)p;
      af.q[1] = *(const v4u*)(p + 8);
#pragma unroll
      for (int dt = 0; dt < 4; ++dt)
        acc[et][dt] = wmma_bf16(af, bf[dt], acc[et][dt]);
    }
  }

  const int slot = (w & 3) * 4096;
  if (w < 4) {
#pragma unroll
    for (int et = 0; et < 4; ++et)
#pragma unroll
      for (int dt = 0; dt < 4; ++dt)
#pragma unroll
        for (int v = 0; v < 8; ++v) {
          const int e = et * 16 + v + 8 * h, d = dt * 16 + r;
          sred[slot + e * 64 + d] = acc[et][dt][v];
        }
  }
  __syncthreads();
  if (w >= 4) {
#pragma unroll
    for (int et = 0; et < 4; ++et)
#pragma unroll
      for (int dt = 0; dt < 4; ++dt)
#pragma unroll
        for (int v = 0; v < 8; ++v) {
          const int e = et * 16 + v + 8 * h, d = dt * 16 + r;
          atomicAdd(&sred[slot + e * 64 + d], acc[et][dt][v]);
        }
  }
  __syncthreads();
  for (int i = t; i < 4096; i += 256) {
    const float s = sred[i] + sred[4096 + i] + sred[8192 + i] + sred[12288 + i];
    const int e = i >> 6, d = i & 63;
    MTb[((size_t)bh * DD + d) * DD + e] = f2bf(s * 0.125f);  // fold 1/sqrt(D)
  }
}

// ---------------------------------------------------------------------------
// K3: O = q @ M  -> Ob [B,N,768] bf16.  Tiles staged by the Tensor Data Mover
// (D# descriptors with hardware LDS padding 128B->144B rows) when available.
// ---------------------------------------------------------------------------
__global__ __launch_bounds__(256) void qm_gemm(
    const unsigned short* __restrict__ Qb,   // [B,H,N,64]
    const unsigned short* __restrict__ MTb,  // [B,H,64(d),64(e)]
    unsigned short* __restrict__ Ob)         // [B,N,768]
{
  __shared__ unsigned short ldsQ[128 * 72];
  __shared__ unsigned short ldsM[64 * 72];
  const int bh = blockIdx.y, b = bh / HH, head = bh % HH;
  const int n0 = blockIdx.x * 128;
  const int t = threadIdx.x, lane = t & 31, w = t >> 5;
  const unsigned short* Qp = Qb + (size_t)bh * NN * DD;
  const unsigned short* Mp = MTb + (size_t)bh * DD * DD;

#if HAVE_TDM
  if (t == 0) {
    tdm_load_2d_pad((unsigned)(size_t)ldsM, Mp, DD, DD, DD, DD, DD);
    tdm_load_2d_pad((unsigned)(size_t)ldsQ, Qp + (size_t)n0 * DD,
                    DD, NN, DD, 128, DD);
    __builtin_amdgcn_s_wait_tensorcnt(0);
  }
  __syncthreads();
#else
  { // stage M head (64x64) into LDS, padded stride 72
    const int row = t >> 2, part = t & 3;
    const v4u* s = (const v4u*)(Mp + row * 64 + part * 16);
    v4u m0 = s[0], m1 = s[1];
    v4u* d = (v4u*)(ldsM + row * 72 + part * 16);
    d[0] = m0; d[1] = m1;
  }
  { // stage Q tile (128 rows x 64)
    const int row = t >> 1, part = t & 1;
    const v4u* s = (const v4u*)(Qp + (size_t)(n0 + row) * DD + part * 32);
    v4u q0 = s[0], q1 = s[1], q2 = s[2], q3 = s[3];
    v4u* d = (v4u*)(ldsQ + row * 72 + part * 32);
    d[0] = q0; d[1] = q1; d[2] = q2; d[3] = q3;
  }
  __syncthreads();
#endif

  const int h = lane >> 4, r = lane & 15;
  v8f acc[4] = {};
#pragma unroll
  for (int k0 = 0; k0 < 64; k0 += 32) {
    ABFrag af;
    const unsigned short* p = ldsQ + (w * 16 + r) * 72 + k0 + 16 * h;
    af.q[0] = *(const v4u*)p;
    af.q[1] = *(const v4u*)(p + 8);
#pragma unroll
    for (int dt = 0; dt < 4; ++dt) {
      ABFrag bfr;
      const unsigned short* pb = ldsM + (dt * 16 + r) * 72 + k0 + 16 * h;
      bfr.q[0] = *(const v4u*)pb;
      bfr.q[1] = *(const v4u*)(pb + 8);
      acc[dt] = wmma_bf16(af, bfr, acc[dt]);
    }
  }
#pragma unroll
  for (int dt = 0; dt < 4; ++dt)
#pragma unroll
    for (int v = 0; v < 8; ++v) {
      const int n = n0 + w * 16 + v + 8 * h;
      const int d = dt * 16 + r;
      Ob[((size_t)b * NN + n) * CC + head * DD + d] = f2bf(acc[dt][v]);
    }
}

// ---------------------------------------------------------------------------
// K4: proj GEMM  out = Ob @ w_proj + b_proj   (f32 output)
// ---------------------------------------------------------------------------
__global__ __launch_bounds__(256) void proj_gemm(
    const unsigned short* __restrict__ Ob,      // [8192][768] bf16
    const unsigned short* __restrict__ WprojT,  // [768][768] bf16 (transposed)
    const float* __restrict__ b_proj,
    float* __restrict__ out)                    // [8192][768] f32
{
  __shared__ unsigned short ldsA[2 * TBUF];
  __shared__ unsigned short ldsB[2 * TBUF];
  const int m0 = blockIdx.x * 128, n0 = blockIdx.y * 128;
  v8f acc[4][2] = {};
  gemm128x128_bf16(Ob, CC, WprojT, CC, m0, n0, CC, ldsA, ldsB, acc);

  const int lane = threadIdx.x & 31, w = threadIdx.x >> 5;
  const int wm = w & 1, wn = w >> 1;
  const int h = lane >> 4, r = lane & 15;
#pragma unroll
  for (int mt = 0; mt < 4; ++mt) {
#pragma unroll
    for (int nt = 0; nt < 2; ++nt) {
      const int col = n0 + wn * 32 + nt * 16 + r;
      const float bias = b_proj[col];
#pragma unroll
      for (int v = 0; v < 8; ++v) {
        const int row = m0 + wm * 64 + mt * 16 + v + 8 * h;
        out[(size_t)row * CC + col] = acc[mt][nt][v] + bias;
      }
    }
  }
}

// ---------------------------------------------------------------------------
// Launch.  Workspace layout (bytes, 256-aligned), total ~64.7 MB.
// ---------------------------------------------------------------------------
extern "C" void kernel_launch(void* const* d_in, const int* in_sizes, int n_in,
                              void* d_out, int out_size, void* d_ws, size_t ws_size,
                              hipStream_t stream) {
  const float* x      = (const float*)d_in[0];   // [2,4096,768]
  const float* w_qkv  = (const float*)d_in[1];   // [768,2304]
  const float* b_qkv  = (const float*)d_in[2];   // [2304]
  const float* w_proj = (const float*)d_in[3];   // [768,768]
  const float* b_proj = (const float*)d_in[4];   // [768]
  float* out = (float*)d_out;
  char* ws = (char*)d_ws;

  size_t off = 0;
  auto take = [&](size_t bytes) { size_t o = off; off = (off + bytes + 255) & ~(size_t)255; return o; };
  unsigned short* Xb     = (unsigned short*)(ws + take((size_t)ROWS * CC * 2));
  unsigned short* WqkvT  = (unsigned short*)(ws + take((size_t)TC * CC * 2));
  unsigned short* WprojT = (unsigned short*)(ws + take((size_t)CC * CC * 2));
  unsigned short* Qb     = (unsigned short*)(ws + take((size_t)ROWS * CC * 2));
  unsigned short* Kt     = (unsigned short*)(ws + take((size_t)ROWS * CC * 2));
  unsigned short* Vt     = (unsigned short*)(ws + take((size_t)ROWS * CC * 2));
  unsigned short* MTb    = (unsigned short*)(ws + take((size_t)BB * HH * DD * DD * 2));
  unsigned short* Ob     = (unsigned short*)(ws + take((size_t)ROWS * CC * 2));

  const int nx  = ROWS * CC;         // 6,291,456
  const int nwq = CC * TC;           // 1,769,472
  const int nwp = CC * CC;           //   589,824
  cvt_f32_bf16<<<(nx + 255) / 256, 256, 0, stream>>>(x, Xb, nx);
  cvt_transpose_bf16<<<(nwq + 255) / 256, 256, 0, stream>>>(w_qkv, WqkvT, CC, TC);
  cvt_transpose_bf16<<<(nwp + 255) / 256, 256, 0, stream>>>(w_proj, WprojT, CC, CC);

  qkv_gemm<<<dim3(ROWS / 128, TC / 128), 256, 0, stream>>>(Xb, WqkvT, b_qkv, Qb, Kt, Vt);
  kv_outer<<<BB * HH, 256, 0, stream>>>(Kt, Vt, MTb);
  qm_gemm<<<dim3(NN / 128, BB * HH), 256, 0, stream>>>(Qb, MTb, Ob);
  proj_gemm<<<dim3(ROWS / 128, CC / 128), 256, 0, stream>>>(Ob, WprojT, b_proj, out);
}
